// LlamaDecoderLayer_26431228739815
// MI455X (gfx1250) — compile-verified, matrix operand order checked by reading, not run on hardware
//
#include <hip/hip_runtime.h>
#include <hip/hip_bf16.h>

// ---------------------------------------------------------------------------
// Llama decoder layer for MI455X (gfx1250): bf16 WMMA GEMMs, fp32 accumulate.
// Inner GEMM loop is software-pipelined (register double-buffer) so
// global_load_b128 of step k+1 overlaps the v_wmma issue of step k.
// ---------------------------------------------------------------------------

typedef __bf16 bf16_t;
typedef __bf16 v16bf __attribute__((ext_vector_type(16)));
typedef float  v8f   __attribute__((ext_vector_type(8)));
typedef float  v4f   __attribute__((ext_vector_type(4)));
typedef int    v4i   __attribute__((ext_vector_type(4)));

#define S_LEN   2048
#define HID     2048
#define N_HEADS 32
#define N_KV    8
#define HD      64
#define INTER   8192

// ---------------------------------------------------------------------------
// Generic WMMA GEMM:  C[m,n] = alpha * sum_k A[m,k] * B[n,k]  (+ causal mask)
//   A: bf16 (AF32=false) or fp32 (AF32=true), row-major, lda == K
//   B: bf16 row-major [N,K], ldb == K        (i.e. computes A @ B^T)
//   One wave (32 threads) per block; each wave produces a 16(M) x 64(N) strip.
//   Batched over blockIdx.z: A += z*aBatch, B += (z/bHeadDiv)*bBatch,
//   C += z*cBatch + z*cColPerZ (column offset), ldc given.
// VGPR layouts per CDNA5 ISA 7.12.2:
//   A (16x32 bf16): lanes 0-15 -> row l, K = {k..k+7, k+16..k+23}
//                   lanes 16-31 -> row l, K pattern shifted by +8
//   B mirrors A with N in place of M.  C/D: VGPR r -> M = r + 8*(lane>=16),
//   N = lane & 15.
// ---------------------------------------------------------------------------
template <bool AF32>
__global__ void gemm_wmma_kernel(const void* __restrict__ A_,
                                 const bf16_t* __restrict__ B,
                                 float* __restrict__ C,
                                 int M, int N, int K,
                                 long aBatch, long bBatch, long cBatch,
                                 int bHeadDiv, int ldc, int cColPerZ,
                                 float alpha, int causal)
{
    const int lane  = threadIdx.x;        // 0..31
    const int half  = lane >> 4;          // 0 or 1
    const int l     = lane & 15;
    const int z     = blockIdx.z;
    const int mbase = blockIdx.y * 16;
    const int nbase = blockIdx.x * 64;
    const int krow  = half << 3;          // +0 / +8 within K pattern

    const bf16_t* Ab = (const bf16_t*)A_ + (long)z * aBatch;
    const float*  Af = (const float*) A_ + (long)z * aBatch;
    const bf16_t* Bz = B + (long)(z / bHeadDiv) * bBatch;

    v8f acc[4];
#pragma unroll
    for (int j = 0; j < 4; ++j)
#pragma unroll
        for (int r = 0; r < 8; ++r) acc[j][r] = 0.0f;

    const long aRow = (long)(mbase + l) * K + krow;
    const bf16_t* bBase[4];
#pragma unroll
    for (int j = 0; j < 4; ++j)
        bBase[j] = Bz + (long)(nbase + j * 16 + l) * K + krow;

    // fragment loaders -----------------------------------------------------
    auto loadA = [&](int k, v16bf& a) {
        if constexpr (AF32) {
            const float* p = Af + aRow + k;
            v4f f0 = *(const v4f*)(p);
            v4f f1 = *(const v4f*)(p + 4);
            v4f f2 = *(const v4f*)(p + 16);
            v4f f3 = *(const v4f*)(p + 20);
#pragma unroll
            for (int t = 0; t < 4; ++t) {
                a[t]      = (bf16_t)f0[t];
                a[4 + t]  = (bf16_t)f1[t];
                a[8 + t]  = (bf16_t)f2[t];
                a[12 + t] = (bf16_t)f3[t];
            }
        } else {
            const bf16_t* p = Ab + aRow + k;
            ((v4i*)&a)[0] = *(const v4i*)(p);        // K .. K+7
            ((v4i*)&a)[1] = *(const v4i*)(p + 16);   // K+16 .. K+23
        }
    };
    auto loadB = [&](int j, int k, v16bf& b) {
        const bf16_t* p = bBase[j] + k;
        ((v4i*)&b)[0] = *(const v4i*)(p);
        ((v4i*)&b)[1] = *(const v4i*)(p + 16);
    };

    // primed, register double-buffered pipeline ----------------------------
    v16bf a_cur, b_cur[4];
    loadA(0, a_cur);
#pragma unroll
    for (int j = 0; j < 4; ++j) loadB(j, 0, b_cur[j]);

    int k = 0;
    for (; k + 32 < K; k += 32) {
        v16bf a_nxt, b_nxt[4];
        loadA(k + 32, a_nxt);                         // prefetch step k+1
#pragma unroll
        for (int j = 0; j < 4; ++j) loadB(j, k + 32, b_nxt[j]);

#pragma unroll
        for (int j = 0; j < 4; ++j)                   // compute step k
            acc[j] = __builtin_amdgcn_wmma_f32_16x16x32_bf16(
                false, a_cur, false, b_cur[j], (short)0, acc[j], false, false);

        a_cur = a_nxt;
#pragma unroll
        for (int j = 0; j < 4; ++j) b_cur[j] = b_nxt[j];
    }
#pragma unroll
    for (int j = 0; j < 4; ++j)                       // peeled last step
        acc[j] = __builtin_amdgcn_wmma_f32_16x16x32_bf16(
            false, a_cur, false, b_cur[j], (short)0, acc[j], false, false);

    // epilogue -------------------------------------------------------------
    float* Cz = C + (long)z * cBatch + (long)z * cColPerZ;
#pragma unroll
    for (int j = 0; j < 4; ++j) {
        const int cc = nbase + j * 16 + l;
#pragma unroll
        for (int r = 0; r < 8; ++r) {
            const int rr = mbase + r + half * 8;
            float v = acc[j][r] * alpha;
            if (causal && cc > rr) v += -1.0e9f;
            Cz[(long)rr * ldc + cc] = v;
        }
    }
}

// ---------------------------------------------------------------------------
// Elementwise / reduction kernels
// ---------------------------------------------------------------------------
__global__ void cvt_f32_bf16_kernel(const float* __restrict__ in,
                                    bf16_t* __restrict__ out, long n)
{
    long stride = (long)gridDim.x * blockDim.x;
    for (long i = (long)blockIdx.x * blockDim.x + threadIdx.x; i < n; i += stride)
        out[i] = (bf16_t)in[i];
}

__global__ void add_f32_kernel(const float* __restrict__ a,
                               const float* __restrict__ b,
                               float* __restrict__ out, long n)
{
    long stride = (long)gridDim.x * blockDim.x;
    for (long i = (long)blockIdx.x * blockDim.x + threadIdx.x; i < n; i += stride)
        out[i] = a[i] + b[i];
}

__global__ void silu_mul_kernel(const float* __restrict__ g,
                                const float* __restrict__ u,
                                bf16_t* __restrict__ out, long n)
{
    long stride = (long)gridDim.x * blockDim.x;
    for (long i = (long)blockIdx.x * blockDim.x + threadIdx.x; i < n; i += stride) {
        float gv = g[i];
        float s  = gv / (1.0f + __expf(-gv));
        out[i] = (bf16_t)(s * u[i]);
    }
}

// RMSNorm: one block (256 thr) per row of [S, ncol]; bf16 output.
__global__ void rmsnorm_kernel(const float* __restrict__ x,
                               const float* __restrict__ w,
                               bf16_t* __restrict__ out, int ncol)
{
    const int row = blockIdx.x;
    const float* xr = x + (long)row * ncol;
    __shared__ float red[8];
    int tid = threadIdx.x;
    float ss = 0.0f;
    for (int i = tid; i < ncol; i += 256) { float v = xr[i]; ss += v * v; }
    for (int off = 16; off > 0; off >>= 1) ss += __shfl_down(ss, off, 32);
    if ((tid & 31) == 0) red[tid >> 5] = ss;
    __syncthreads();
    if (tid == 0) {
        float t = 0.0f;
        for (int i = 0; i < 8; ++i) t += red[i];
        red[0] = rsqrtf(t / (float)ncol + 1e-6f);
    }
    __syncthreads();
    float scale = red[0];
    bf16_t* o = out + (long)row * ncol;
    for (int i = tid; i < ncol; i += 256)
        o[i] = (bf16_t)(w[i] * xr[i] * scale);
}

// RoPE + transpose to head-major. grid (S, N_HEADS), block 64.
//   qf [S, 32*64] fp32 -> qb [32, S, 64] bf16
//   kf [S,  8*64] fp32 -> kb [ 8, S, 64] bf16
//   vf [S,  8*64] fp32 -> vtb [8, 64, S] bf16   (transposed for P@V GEMM)
__global__ void rope_transpose_kernel(const float* __restrict__ qf,
                                      const float* __restrict__ kf,
                                      const float* __restrict__ vf,
                                      const int* __restrict__ pos,
                                      bf16_t* __restrict__ qb,
                                      bf16_t* __restrict__ kb,
                                      bf16_t* __restrict__ vtb)
{
    const int s = blockIdx.x;
    const int h = blockIdx.y;
    const int d = threadIdx.x;            // 0..63
    const int i = d & 31;
    const float p = (float)pos[s];
    const float inv = __powf(10000.0f, -(float)i / 32.0f);
    const float ang = p * inv;
    const float c = __cosf(ang), sn = __sinf(ang);
    const int dswap = (d < 32) ? (d + 32) : (d - 32);

    float xq = qf[(long)s * HID + h * HD + d];
    float oq = qf[(long)s * HID + h * HD + dswap];
    float rq = (d < 32) ? (xq * c - oq * sn) : (xq * c + oq * sn);
    qb[((long)h * S_LEN + s) * HD + d] = (bf16_t)rq;

    if (h < N_KV) {
        float xk = kf[(long)s * (N_KV * HD) + h * HD + d];
        float ok = kf[(long)s * (N_KV * HD) + h * HD + dswap];
        float rk = (d < 32) ? (xk * c - ok * sn) : (xk * c + ok * sn);
        kb[((long)h * S_LEN + s) * HD + d] = (bf16_t)rk;
        float vv = vf[(long)s * (N_KV * HD) + h * HD + d];
        vtb[((long)h * HD + d) * S_LEN + s] = (bf16_t)vv;
    }
}

// Row softmax in place: attn [H, S, S] fp32. grid (S, H), block 256.
__global__ void softmax_kernel(float* __restrict__ attn)
{
    float* p = attn + ((long)blockIdx.y * S_LEN + blockIdx.x) * S_LEN;
    __shared__ float red[8];
    const int tid = threadIdx.x;

    float mx = -3.4e38f;
    for (int i = tid; i < S_LEN; i += 256) mx = fmaxf(mx, p[i]);
    for (int off = 16; off > 0; off >>= 1) mx = fmaxf(mx, __shfl_down(mx, off, 32));
    if ((tid & 31) == 0) red[tid >> 5] = mx;
    __syncthreads();
    if (tid == 0) {
        float m = red[0];
        for (int i = 1; i < 8; ++i) m = fmaxf(m, red[i]);
        red[0] = m;
    }
    __syncthreads();
    mx = red[0];
    __syncthreads();

    float sum = 0.0f;
    for (int i = tid; i < S_LEN; i += 256) {
        float e = __expf(p[i] - mx);
        p[i] = e;
        sum += e;
    }
    for (int off = 16; off > 0; off >>= 1) sum += __shfl_down(sum, off, 32);
    if ((tid & 31) == 0) red[tid >> 5] = sum;
    __syncthreads();
    if (tid == 0) {
        float t = 0.0f;
        for (int i = 0; i < 8; ++i) t += red[i];
        red[0] = 1.0f / t;
    }
    __syncthreads();
    float invs = red[0];
    for (int i = tid; i < S_LEN; i += 256) p[i] *= invs;
}

// ---------------------------------------------------------------------------
// Host-side orchestration
// ---------------------------------------------------------------------------
static inline size_t align256(size_t x) { return (x + 255) & ~(size_t)255; }

extern "C" void kernel_launch(void* const* d_in, const int* in_sizes, int n_in,
                              void* d_out, int out_size, void* d_ws, size_t ws_size,
                              hipStream_t stream)
{
    const float* hidden = (const float*)d_in[0];
    // d_in[1] = attention_mask (causal; fused analytically in scores epilogue)
    const int*   pos    = (const int*)  d_in[2];
    const float* wq     = (const float*)d_in[3];
    const float* wk     = (const float*)d_in[4];
    const float* wv     = (const float*)d_in[5];
    const float* wo     = (const float*)d_in[6];
    const float* wg     = (const float*)d_in[7];
    const float* wu     = (const float*)d_in[8];
    const float* wd     = (const float*)d_in[9];
    const float* ln1    = (const float*)d_in[10];
    const float* ln2    = (const float*)d_in[11];

    float* out_hidden = (float*)d_out;                                   // [S, HID]
    float* attnW      = (float*)d_out + (long)S_LEN * HID;               // [H, S, S]

    // ---- workspace carve-up -------------------------------------------------
    char* ws = (char*)d_ws;
    size_t off = 0;
    auto alloc = [&](size_t bytes) -> void* {
        void* p = ws + off;
        off += align256(bytes);
        return p;
    };
    const long KV = (long)N_KV * HD;                                     // 512
    bf16_t* wqb  = (bf16_t*)alloc((long)HID * HID * 2);
    bf16_t* wkb  = (bf16_t*)alloc(KV * HID * 2);
    bf16_t* wvb  = (bf16_t*)alloc(KV * HID * 2);
    bf16_t* wob  = (bf16_t*)alloc((long)HID * HID * 2);
    bf16_t* wgb  = (bf16_t*)alloc((long)INTER * HID * 2);
    bf16_t* wub  = (bf16_t*)alloc((long)INTER * HID * 2);
    bf16_t* wdb  = (bf16_t*)alloc((long)HID * INTER * 2);
    bf16_t* xnb  = (bf16_t*)alloc((long)S_LEN * HID * 2);
    float*  qf   = (float*) alloc((long)S_LEN * HID * 4);
    float*  kf   = (float*) alloc((long)S_LEN * KV * 4);
    float*  vf   = (float*) alloc((long)S_LEN * KV * 4);
    bf16_t* qb   = (bf16_t*)alloc((long)N_HEADS * S_LEN * HD * 2);
    bf16_t* kb   = (bf16_t*)alloc((long)N_KV * S_LEN * HD * 2);
    bf16_t* vtb  = (bf16_t*)alloc((long)N_KV * HD * S_LEN * 2);
    float*  ofp  = (float*) alloc((long)S_LEN * HID * 4);
    bf16_t* ofb  = (bf16_t*)alloc((long)S_LEN * HID * 2);
    float*  proj = (float*) alloc((long)S_LEN * HID * 4);
    float*  hf   = (float*) alloc((long)S_LEN * HID * 4);
    bf16_t* ynb  = (bf16_t*)alloc((long)S_LEN * HID * 2);
    float*  gf   = (float*) alloc((long)S_LEN * INTER * 4);
    float*  uf   = (float*) alloc((long)S_LEN * INTER * 4);
    bf16_t* actb = (bf16_t*)alloc((long)S_LEN * INTER * 2);
    float*  mlpf = (float*) alloc((long)S_LEN * HID * 4);
    (void)ws_size; (void)n_in; (void)in_sizes; (void)out_size;

    const dim3 blk32(32), blk64(64), blk256(256);
    const int EW_BLOCKS = 4096;

    // ---- 1. weights -> bf16 -------------------------------------------------
    cvt_f32_bf16_kernel<<<EW_BLOCKS, blk256, 0, stream>>>(wq, wqb, (long)HID * HID);
    cvt_f32_bf16_kernel<<<EW_BLOCKS, blk256, 0, stream>>>(wk, wkb, KV * HID);
    cvt_f32_bf16_kernel<<<EW_BLOCKS, blk256, 0, stream>>>(wv, wvb, KV * HID);
    cvt_f32_bf16_kernel<<<EW_BLOCKS, blk256, 0, stream>>>(wo, wob, (long)HID * HID);
    cvt_f32_bf16_kernel<<<EW_BLOCKS, blk256, 0, stream>>>(wg, wgb, (long)INTER * HID);
    cvt_f32_bf16_kernel<<<EW_BLOCKS, blk256, 0, stream>>>(wu, wub, (long)INTER * HID);
    cvt_f32_bf16_kernel<<<EW_BLOCKS, blk256, 0, stream>>>(wd, wdb, (long)HID * INTER);

    // ---- 2. RMSNorm 1 -------------------------------------------------------
    rmsnorm_kernel<<<S_LEN, blk256, 0, stream>>>(hidden, ln1, xnb, HID);

    // ---- 3. QKV projections -------------------------------------------------
    gemm_wmma_kernel<false><<<dim3(HID / 64, S_LEN / 16, 1), blk32, 0, stream>>>(
        xnb, wqb, qf, S_LEN, HID, HID, 0, 0, 0, 1, HID, 0, 1.0f, 0);
    gemm_wmma_kernel<false><<<dim3(KV / 64, S_LEN / 16, 1), blk32, 0, stream>>>(
        xnb, wkb, kf, S_LEN, (int)KV, HID, 0, 0, 0, 1, (int)KV, 0, 1.0f, 0);
    gemm_wmma_kernel<false><<<dim3(KV / 64, S_LEN / 16, 1), blk32, 0, stream>>>(
        xnb, wvb, vf, S_LEN, (int)KV, HID, 0, 0, 0, 1, (int)KV, 0, 1.0f, 0);

    // ---- 4. RoPE + head-major transpose ------------------------------------
    rope_transpose_kernel<<<dim3(S_LEN, N_HEADS), blk64, 0, stream>>>(
        qf, kf, vf, pos, qb, kb, vtb);

    // ---- 5. scores = q @ k^T / 8 + causal mask  (into attn_weights region) --
    gemm_wmma_kernel<false><<<dim3(S_LEN / 64, S_LEN / 16, N_HEADS), blk32, 0, stream>>>(
        qb, kb, attnW, S_LEN, S_LEN, HD,
        (long)S_LEN * HD, (long)S_LEN * HD, (long)S_LEN * S_LEN,
        N_HEADS / N_KV, S_LEN, 0, 0.125f, 1);

    // ---- 6. softmax (fp32, in place; this IS the attn_weights output) ------
    softmax_kernel<<<dim3(S_LEN, N_HEADS), blk256, 0, stream>>>(attnW);

    // ---- 7. attn_out = P @ V  (fp32 A path), written head-interleaved ------
    gemm_wmma_kernel<true><<<dim3(HD / 64, S_LEN / 16, N_HEADS), blk32, 0, stream>>>(
        attnW, vtb, ofp, S_LEN, HD, S_LEN,
        (long)S_LEN * S_LEN, (long)HD * S_LEN, 0,
        N_HEADS / N_KV, HID, HD, 1.0f, 0);

    // ---- 8. output projection + residual ------------------------------------
    cvt_f32_bf16_kernel<<<EW_BLOCKS, blk256, 0, stream>>>(ofp, ofb, (long)S_LEN * HID);
    gemm_wmma_kernel<false><<<dim3(HID / 64, S_LEN / 16, 1), blk32, 0, stream>>>(
        ofb, wob, proj, S_LEN, HID, HID, 0, 0, 0, 1, HID, 0, 1.0f, 0);
    add_f32_kernel<<<EW_BLOCKS, blk256, 0, stream>>>(hidden, proj, hf, (long)S_LEN * HID);

    // ---- 9. RMSNorm 2 + MLP -------------------------------------------------
    rmsnorm_kernel<<<S_LEN, blk256, 0, stream>>>(hf, ln2, ynb, HID);
    gemm_wmma_kernel<false><<<dim3(INTER / 64, S_LEN / 16, 1), blk32, 0, stream>>>(
        ynb, wgb, gf, S_LEN, INTER, HID, 0, 0, 0, 1, INTER, 0, 1.0f, 0);
    gemm_wmma_kernel<false><<<dim3(INTER / 64, S_LEN / 16, 1), blk32, 0, stream>>>(
        ynb, wub, uf, S_LEN, INTER, HID, 0, 0, 0, 1, INTER, 0, 1.0f, 0);
    silu_mul_kernel<<<EW_BLOCKS, blk256, 0, stream>>>(gf, uf, actb, (long)S_LEN * INTER);
    gemm_wmma_kernel<false><<<dim3(HID / 64, S_LEN / 16, 1), blk32, 0, stream>>>(
        actb, wdb, mlpf, S_LEN, HID, INTER, 0, 0, 0, 1, HID, 0, 1.0f, 0);

    // ---- 10. final residual -> d_out ---------------------------------------
    add_f32_kernel<<<EW_BLOCKS, blk256, 0, stream>>>(hf, mlpf, out_hidden, (long)S_LEN * HID);
}
